// ConGAE_sp_5282809774364
// MI455X (gfx1250) — matrix operand order repaired and checked by reading, not compiled, not run on hardware
//
#include <hip/hip_runtime.h>
#include <hip/hip_bf16.h>
#include <math.h>

// ---------------- problem constants ----------------
#define N_NODES 1000
#define NB      64
#define NTOT    (NB * N_NODES)      // 64000 nodes
#define NE      2048000             // edges
#define F_IN    128
#define D1      256
#define D2      64

typedef __attribute__((ext_vector_type(16))) _Float16 v16h;
typedef __attribute__((ext_vector_type(8)))  float    v8f;

// ---------------- WMMA fragment loaders (gfx1250 wave32 layouts) ----------------
// A 16x32 f16: lane l holds row (l&15); element e -> K = (e&7) + 16*(e>>3) + 8*(l>>4)
__device__ __forceinline__ v16h load_a_from_f32(const float* A, int lda, int row,
                                                int ktile, int lane) {
  const int kbase = ktile * 32 + 8 * (lane >> 4);
  const float* p = A + (size_t)row * lda + kbase;
  v16h a;
#pragma unroll
  for (int i = 0; i < 8; ++i) {
    a[i]     = (_Float16)p[i];
    a[8 + i] = (_Float16)p[16 + i];
  }
  return a;
}

__device__ __forceinline__ v16h load_a_from_f16(const _Float16* A, int lda, int row,
                                                int ktile, int lane) {
  const int kbase = ktile * 32 + 8 * (lane >> 4);
  const _Float16* p = A + (size_t)row * lda + kbase;
  v16h a;
#pragma unroll
  for (int i = 0; i < 8; ++i) {
    a[i]     = p[i];
    a[8 + i] = p[16 + i];
  }
  return a;
}

// B 32x16 f16 with B[k][n] = W[n][k] (W row-major [N x K]):
// lane l holds col (l&15); element e -> K = ktile*32 + (e&7) + 16*(e>>3) + 8*(l>>4)
__device__ __forceinline__ v16h load_b_wT(const _Float16* W, int ldw, int nbase,
                                          int ktile, int lane) {
  const _Float16* p = W + (size_t)(nbase + (lane & 15)) * ldw + ktile * 32 + 8 * (lane >> 4);
  v16h b;
#pragma unroll
  for (int i = 0; i < 8; ++i) { b[i] = p[i]; b[8 + i] = p[16 + i]; }
  return b;
}

__device__ __forceinline__ v8f wmma_f16(v16h a, v16h b, v8f c) {
  return __builtin_amdgcn_wmma_f32_16x16x32_f16(false, a, false, b, (short)0, c,
                                                false, false);
}

// ---------------- small utility kernels ----------------
__global__ void cvt_f32_to_f16(const float* __restrict__ in, _Float16* __restrict__ out,
                               int n) {
  int i = blockIdx.x * blockDim.x + threadIdx.x;
  if (i < n) out[i] = (_Float16)in[i];
}

// ---------------- GraphConv message/scatter kernels ----------------
// agg1[dst, f] += x[src, f] * ew   (F_IN = 128, 32 threads/edge, 4 floats/thread)
__global__ void scatter_conv1(const float* __restrict__ x,
                              const long long* __restrict__ src,
                              const long long* __restrict__ dst,
                              const float* __restrict__ ew,
                              float* __restrict__ agg1) {
  long long t = (long long)blockIdx.x * blockDim.x + threadIdx.x;
  int e = (int)(t >> 5);
  if (e >= NE) return;
  int f = (int)(t & 31) * 4;
  long long s = src[e], d = dst[e];
  float w = ew[e];
  const float4 xv = *(const float4*)(x + s * F_IN + f);
  float* o = agg1 + d * F_IN + f;
  unsafeAtomicAdd(o + 0, xv.x * w);
  unsafeAtomicAdd(o + 1, xv.y * w);
  unsafeAtomicAdd(o + 2, xv.z * w);
  unsafeAtomicAdd(o + 3, xv.w * w);
}

// agg2[dst, f] += h[src, f] * ew   (D1 = 256 f16 source, 64 threads/edge)
__global__ void scatter_conv2(const _Float16* __restrict__ h,
                              const long long* __restrict__ src,
                              const long long* __restrict__ dst,
                              const float* __restrict__ ew,
                              float* __restrict__ agg2) {
  long long t = (long long)blockIdx.x * blockDim.x + threadIdx.x;
  int e = (int)(t >> 6);
  if (e >= NE) return;
  int f = (int)(t & 63) * 4;
  long long s = src[e], d = dst[e];
  float w = ew[e];
  const _Float16* hp = h + s * D1 + f;
  float* o = agg2 + d * D1 + f;
#pragma unroll
  for (int i = 0; i < 4; ++i) unsafeAtomicAdd(o + i, (float)hp[i] * w);
}

// ---------------- conv1 GEMM: h = relu(agg1@Wl1^T + x@Wr1^T + bl1 + br1) -> f16 ----------------
// M=64000, K=128, N=256. 8 waves/block, one 16x16 tile per wave.
__global__ __launch_bounds__(256) void conv1_gemm(const float* __restrict__ agg1,
                                                  const _Float16* __restrict__ Wl1h,
                                                  const float* __restrict__ x,
                                                  const _Float16* __restrict__ Wr1h,
                                                  const float* __restrict__ bl1,
                                                  const float* __restrict__ br1,
                                                  _Float16* __restrict__ H) {
  const int lane = threadIdx.x & 31;
  const int wid  = threadIdx.x >> 5;
  const int mt   = blockIdx.x * 8 + wid;   // 4000 m-tiles
  const int nt   = blockIdx.y;             // 16 n-tiles
  const int row  = mt * 16 + (lane & 15);

  v8f c = {};
#pragma unroll
  for (int kt = 0; kt < F_IN / 32; ++kt) {
    v16h a = load_a_from_f32(agg1, F_IN, row, kt, lane);
    v16h b = load_b_wT(Wl1h, F_IN, nt * 16, kt, lane);
    c = wmma_f16(a, b, c);
  }
#pragma unroll
  for (int kt = 0; kt < F_IN / 32; ++kt) {
    v16h a = load_a_from_f32(x, F_IN, row, kt, lane);
    v16h b = load_b_wT(Wr1h, F_IN, nt * 16, kt, lane);
    c = wmma_f16(a, b, c);
  }

  const int col   = nt * 16 + (lane & 15);
  const int rbase = mt * 16 + 8 * (lane >> 4);
  const float bs  = bl1[col] + br1[col];
#pragma unroll
  for (int r = 0; r < 8; ++r) {
    float v = c[r] + bs;
    v = v > 0.0f ? v : 0.0f;
    H[(size_t)(rbase + r) * D1 + col] = (_Float16)v;
  }
}

// ---------------- conv2 GEMM: z = agg2@Wl2^T + h@Wr2^T + bl2 + br2 -> f32 ----------------
// M=64000, K=256, N=64.
__global__ __launch_bounds__(256) void conv2_gemm(const float* __restrict__ agg2,
                                                  const _Float16* __restrict__ Wl2h,
                                                  const _Float16* __restrict__ h,
                                                  const _Float16* __restrict__ Wr2h,
                                                  const float* __restrict__ bl2,
                                                  const float* __restrict__ br2,
                                                  float* __restrict__ Z) {
  const int lane = threadIdx.x & 31;
  const int wid  = threadIdx.x >> 5;
  const int mt   = blockIdx.x * 8 + wid;   // 4000 m-tiles
  const int nt   = blockIdx.y;             // 4 n-tiles
  const int row  = mt * 16 + (lane & 15);

  v8f c = {};
#pragma unroll
  for (int kt = 0; kt < D1 / 32; ++kt) {
    v16h a = load_a_from_f32(agg2, D1, row, kt, lane);
    v16h b = load_b_wT(Wl2h, D1, nt * 16, kt, lane);
    c = wmma_f16(a, b, c);
  }
#pragma unroll
  for (int kt = 0; kt < D1 / 32; ++kt) {
    v16h a = load_a_from_f16(h, D1, row, kt, lane);
    v16h b = load_b_wT(Wr2h, D1, nt * 16, kt, lane);
    c = wmma_f16(a, b, c);
  }

  const int col   = nt * 16 + (lane & 15);
  const int rbase = mt * 16 + 8 * (lane >> 4);
  const float bs  = bl2[col] + br2[col];
#pragma unroll
  for (int r = 0; r < 8; ++r)
    Z[(size_t)(rbase + r) * D2 + col] = c[r] + bs;
}

// ---------------- bottleneck FC: zfc[b,j] = sum_k relu(z[b,k]) * Wfc[j,k] + bfc[j] ----------------
__global__ __launch_bounds__(256) void fc_reduce(const float* __restrict__ Z,
                                                 const float* __restrict__ Wfc,
                                                 const float* __restrict__ bfc,
                                                 float* __restrict__ Zfc) {
  __shared__ float sbuf[256];
  const int j = blockIdx.x;   // 64
  const int b = blockIdx.y;   // 64
  const float* zr = Z + (size_t)b * (N_NODES * D2);
  const float* wr = Wfc + (size_t)j * (N_NODES * D2);
  float acc = 0.0f;
  for (int k = threadIdx.x; k < N_NODES * D2; k += 256) {
    float zv = zr[k];
    zv = zv > 0.0f ? zv : 0.0f;
    acc += zv * wr[k];
  }
  sbuf[threadIdx.x] = acc;
  __syncthreads();
  for (int s = 128; s > 0; s >>= 1) {
    if (threadIdx.x < s) sbuf[threadIdx.x] += sbuf[threadIdx.x + s];
    __syncthreads();
  }
  if (threadIdx.x == 0) Zfc[b * D2 + j] = sbuf[0] + bfc[j];
}

// ---------------- expansion FC: z2[b*64000+m] = relu(sum_k zfc[b,k]*Wfc2[m,k] + bfc2[m]) -> f16 ----------------
__global__ __launch_bounds__(256) void fc2_relu(const float* __restrict__ Zfc,
                                                const float* __restrict__ Wfc2,
                                                const float* __restrict__ bfc2,
                                                _Float16* __restrict__ Z2h) {
  __shared__ float sz[D2];
  const int b = blockIdx.y;
  if (threadIdx.x < D2) sz[threadIdx.x] = Zfc[b * D2 + threadIdx.x];
  __syncthreads();
  const int m = blockIdx.x * blockDim.x + threadIdx.x;   // 0..63999
  const float* w = Wfc2 + (size_t)m * D2;
  float acc = bfc2[m];
#pragma unroll 8
  for (int k = 0; k < D2; ++k) acc += sz[k] * w[k];
  acc = acc > 0.0f ? acc : 0.0f;
  Z2h[(size_t)b * (N_NODES * D2) + m] = (_Float16)acc;
}

// ---------------- fused edge decoder (LDS-staged weights) ----------------
// A[m, 0..63] = z2[src[m]], A[m, 64..127] = z2[dst[m]]; 16 edges per wave.
// pred = sigmoid( relu(A @ Wd1^T + bd1) @ Wd2^T + bd2 )
// Wd1 (64 KB) staged into LDS once per workgroup, row-padded by 8 halves
// (row stride 272 B = 68 banks -> 4-bank lane offset -> conflict-free b128 reads).
#define LDW (2 * D2 + 8)   // 136 halves per LDS row

__global__ __launch_bounds__(256) void edge_decoder(const _Float16* __restrict__ Z2h,
                                                    const long long* __restrict__ src,
                                                    const long long* __restrict__ dst,
                                                    const _Float16* __restrict__ Wd1h,
                                                    const float* __restrict__ bd1,
                                                    const float* __restrict__ Wd2,
                                                    const float* __restrict__ bd2,
                                                    float* __restrict__ out) {
  __shared__ _Float16 sW[D1 * LDW];   // 256 * 136 * 2 = 69,632 B
  __shared__ float    sBd1[D1];
  __shared__ float    sWd2[D1];

  // cooperative stage: 4096 x 16B chunks, 16 per thread
  for (int idx = threadIdx.x; idx < D1 * (2 * D2 / 8); idx += 256) {
    const int rowi = idx >> 4;        // weight row (output column n)
    const int ch   = idx & 15;        // 8-half chunk within the row
    const uint4 v = *(const uint4*)(Wd1h + (size_t)rowi * (2 * D2) + ch * 8);
    *(uint4*)(&sW[rowi * LDW + ch * 8]) = v;
  }
  for (int i = threadIdx.x; i < D1; i += 256) {
    sBd1[i] = bd1[i];
    sWd2[i] = Wd2[i];
  }
  __syncthreads();

  const int lane = threadIdx.x & 31;
  const int wid  = threadIdx.x >> 5;
  const int et   = blockIdx.x * 8 + wid;   // 128000 edge-tiles
  const int row  = et * 16 + (lane & 15);
  const long long s = src[row];
  const long long d = dst[row];

  // load the 4 K-tiles of A (kt 0,1 from src feature vector; kt 2,3 from dst)
  v16h a[4];
#pragma unroll
  for (int kt = 0; kt < 4; ++kt) {
    const long long node = (kt < 2) ? s : d;
    const _Float16* p = Z2h + node * D2 + (kt & 1) * 32 + 8 * (lane >> 4);
#pragma unroll
    for (int i = 0; i < 8; ++i) {
      a[kt][i]     = p[i];
      a[kt][8 + i] = p[16 + i];
    }
  }

  float acc[8];
#pragma unroll
  for (int r = 0; r < 8; ++r) acc[r] = 0.0f;

#pragma unroll 4
  for (int nt = 0; nt < D1 / 16; ++nt) {
    v8f c = {};
#pragma unroll
    for (int kt = 0; kt < 4; ++kt) {
      // B fragment from LDS: lane covers col (lane&15), K = (e&7) + 16*(e>>3) + 8*(lane>>4)
      const _Float16* p =
          &sW[(nt * 16 + (lane & 15)) * LDW + kt * 32 + 8 * (lane >> 4)];
      v16h b;
#pragma unroll
      for (int i = 0; i < 8; ++i) { b[i] = p[i]; b[8 + i] = p[16 + i]; }
      c = wmma_f16(a[kt], b, c);
    }
    const int col = nt * 16 + (lane & 15);
    const float bb = sBd1[col];
    const float w2 = sWd2[col];
#pragma unroll
    for (int r = 0; r < 8; ++r) {
      float v = c[r] + bb;
      v = v > 0.0f ? v : 0.0f;
      acc[r] += v * w2;
    }
  }

  // reduce across the 16 lanes of each half (different columns, same rows)
#pragma unroll
  for (int r = 0; r < 8; ++r) {
#pragma unroll
    for (int off = 8; off > 0; off >>= 1)
      acc[r] += __shfl_xor(acc[r], off, 16);
  }

  if ((lane & 15) == 0) {
    const float b2 = bd2[0];
    const int base = et * 16 + (lane >> 4) * 8;
#pragma unroll
    for (int r = 0; r < 8; ++r) {
      float p = acc[r] + b2;
      out[base + r] = 1.0f / (1.0f + __expf(-p));
    }
  }
}

// ---------------- host-side orchestration ----------------
extern "C" void kernel_launch(void* const* d_in, const int* in_sizes, int n_in,
                              void* d_out, int out_size, void* d_ws, size_t ws_size,
                              hipStream_t stream) {
  const float*      x    = (const float*)d_in[0];
  const long long*  ei   = (const long long*)d_in[1];
  const float*      ea   = (const float*)d_in[2];
  const float*      Wl1  = (const float*)d_in[3];
  const float*      bl1  = (const float*)d_in[4];
  const float*      Wr1  = (const float*)d_in[5];
  const float*      br1  = (const float*)d_in[6];
  const float*      Wl2  = (const float*)d_in[7];
  const float*      bl2  = (const float*)d_in[8];
  const float*      Wr2  = (const float*)d_in[9];
  const float*      br2  = (const float*)d_in[10];
  const float*      Wfc  = (const float*)d_in[11];
  const float*      bfc  = (const float*)d_in[12];
  const float*      Wfc2 = (const float*)d_in[13];
  const float*      bfc2 = (const float*)d_in[14];
  const float*      Wd1  = (const float*)d_in[15];
  const float*      bd1  = (const float*)d_in[16];
  const float*      Wd2  = (const float*)d_in[17];
  const float*      bd2  = (const float*)d_in[18];

  const long long* src = ei;
  const long long* dst = ei + NE;

  // workspace layout (256B aligned)
  char* ws = (char*)d_ws;
  size_t off = 0;
  auto alloc = [&](size_t bytes) {
    void* p = ws + off;
    off += (bytes + 255) & ~(size_t)255;
    return p;
  };
  float*     agg1 = (float*)alloc((size_t)NTOT * F_IN * 4);     // 32.8 MB
  float*     agg2 = (float*)alloc((size_t)NTOT * D1 * 4);       // 65.5 MB
  _Float16*  H    = (_Float16*)alloc((size_t)NTOT * D1 * 2);    // 32.8 MB
  float*     Z    = (float*)alloc((size_t)NTOT * D2 * 4);       // 16.4 MB
  float*     Zfc  = (float*)alloc((size_t)NB * D2 * 4);
  _Float16*  Z2h  = (_Float16*)alloc((size_t)NTOT * D2 * 2);    // 8.2 MB
  _Float16*  Wl1h = (_Float16*)alloc((size_t)D1 * F_IN * 2);
  _Float16*  Wr1h = (_Float16*)alloc((size_t)D1 * F_IN * 2);
  _Float16*  Wl2h = (_Float16*)alloc((size_t)D2 * D1 * 2);
  _Float16*  Wr2h = (_Float16*)alloc((size_t)D2 * D1 * 2);
  _Float16*  Wd1h = (_Float16*)alloc((size_t)D1 * 2 * D2 * 2);

  // 1) convert weights to f16
  auto cvt = [&](const float* in, _Float16* o, int n) {
    cvt_f32_to_f16<<<(n + 255) / 256, 256, 0, stream>>>(in, o, n);
  };
  cvt(Wl1, Wl1h, D1 * F_IN);
  cvt(Wr1, Wr1h, D1 * F_IN);
  cvt(Wl2, Wl2h, D2 * D1);
  cvt(Wr2, Wr2h, D2 * D1);
  cvt(Wd1, Wd1h, D1 * 2 * D2);

  // 2) zero atomic accumulation buffers (stream memset: graph-capturable)
  hipMemsetAsync(agg1, 0, (size_t)NTOT * F_IN * 4, stream);
  hipMemsetAsync(agg2, 0, (size_t)NTOT * D1 * 4, stream);

  // 3) conv1 scatter: E*32 threads
  {
    long long threads = (long long)NE * 32;
    scatter_conv1<<<(unsigned)((threads + 255) / 256), 256, 0, stream>>>(x, src, dst, ea, agg1);
  }

  // 4) conv1 GEMM + relu -> H (f16)
  conv1_gemm<<<dim3(NTOT / 16 / 8, D1 / 16), 256, 0, stream>>>(agg1, Wl1h, x, Wr1h, bl1, br1, H);

  // 5) conv2 scatter: E*64 threads
  {
    long long threads = (long long)NE * 64;
    scatter_conv2<<<(unsigned)((threads + 255) / 256), 256, 0, stream>>>(H, src, dst, ea, agg2);
  }

  // 6) conv2 GEMM -> Z (f32)
  conv2_gemm<<<dim3(NTOT / 16 / 8, D2 / 16), 256, 0, stream>>>(agg2, Wl2h, H, Wr2h, bl2, br2, Z);

  // 7) bottleneck FC (relu fused on input)
  fc_reduce<<<dim3(D2, NB), 256, 0, stream>>>(Z, Wfc, bfc, Zfc);

  // 8) expansion FC + relu -> Z2h (f16)
  fc2_relu<<<dim3((N_NODES * D2) / 256, NB), 256, 0, stream>>>(Zfc, Wfc2, bfc2, Z2h);

  // 9) fused edge decoder (LDS-staged Wd1) -> sigmoid predictions
  edge_decoder<<<NE / 16 / 8, 256, 0, stream>>>(Z2h, src, dst, Wd1h, bd1, Wd2, bd2,
                                                (float*)d_out);
}
